// MultiHeadAttention_24859270709683
// MI455X (gfx1250) — matrix operand order repaired and checked by reading, not compiled, not run on hardware
//
#include <hip/hip_runtime.h>
#include <hip/hip_bf16.h>
#include <math.h>

// ---------------------------------------------------------------------------
// Multi-head attention forward, MI455X (gfx1250), wave32, WMMA bf16->f32,
// double-buffered Tensor-Data-Mover pipeline for bf16 tiles.
//   D_MODEL=768, H=12, d_k=64, B=4, S=2048 -> M = B*S = 8192 rows.
// ---------------------------------------------------------------------------

#define D_MODEL   768
#define NUM_HEADS 12
#define D_K       64
#define BATCH     4
#define SEQ       2048
#define MROWS     (BATCH * SEQ)   // 8192

typedef __attribute__((ext_vector_type(16))) __bf16 bf16x16;
typedef __attribute__((ext_vector_type(8)))  __bf16 bf16x8;
typedef __attribute__((ext_vector_type(4)))  __bf16 bf16x4;
typedef __attribute__((ext_vector_type(8)))  float  f32x8;
typedef __attribute__((ext_vector_type(4)))  float  f32x4;
typedef __attribute__((ext_vector_type(4)))  unsigned int u32x4;
typedef __attribute__((ext_vector_type(8)))  int    i32x8;
typedef __attribute__((ext_vector_type(4)))  int    i32x4;

__device__ __forceinline__ __bf16 f2bf(float f) { return (__bf16)f; }
__device__ __forceinline__ bf16x4 cvt4(f32x4 v) {
    return __builtin_convertvector(v, bf16x4);
}

// 32-bit LDS byte offset of a __shared__ object (addrspace(3) addresses are
// zero-based LDS offsets).
__device__ __forceinline__ unsigned lds_off(const void* p) {
    return (unsigned)(__SIZE_TYPE__)(const __attribute__((address_space(3))) char*)p;
}

// ---------------------------------------------------------------------------
// Tensor Data Mover: 2-D bf16 tile  global -> LDS, with HW row padding so a
// 64-elem (128B) tile row lands in a 144B LDS row ([.][72] arrays below).
// D# per CDNA5 ISA 8.3/8.4:  group0 = {count/flags, lds_addr, global_addr},
// group1 = {pad/data_size, dims, tile dims, strides}.  2-D: groups 2/3 unused.
// Issued by one wave (EXEC-independent); completion via s_wait_tensorcnt.
// ---------------------------------------------------------------------------
__device__ __forceinline__ void tdm_load_2d_bf16(
    unsigned lds_byte, const void* gptr,
    unsigned tile_w, unsigned tile_h, unsigned stride_elems)
{
    unsigned long long ga = (unsigned long long)gptr;
    u32x4 g0;
    g0.x = 1u;                                          // count=1 (valid), user mode
    g0.y = lds_byte;                                    // LDS destination (bytes)
    g0.z = (unsigned)(ga & 0xFFFFFFFFull);              // global_addr[31:0]
    g0.w = (unsigned)((ga >> 32) & 0x1FFFFFFull)        // global_addr[56:32]
         | (2u << 30);                                  // type = 2 ("image")
    i32x8 g1;
    // pad_amount=3 (4 DWORDs=16B) every pad_interval=4 (32 DWORDs=128B),
    // pad_enable=1, data_size=1 (2 bytes), workgroup_mask=0 (not clustered)
    g1[0] = (int)((3u << 25) | (4u << 22) | (1u << 20) | (1u << 16));
    g1[1] = (int)((tile_w & 0xFFFFu) << 16);            // tensor_dim0[15:0]
    g1[2] = (int)(((tile_w >> 16) & 0xFFFFu) |          // tensor_dim0[31:16]
                  ((tile_h & 0xFFFFu) << 16));          // tensor_dim1[15:0]
    g1[3] = (int)(((tile_h >> 16) & 0xFFFFu) |          // tensor_dim1[31:16]
                  ((tile_w & 0xFFFFu) << 16));          // tile_dim0
    g1[4] = (int)(tile_h & 0xFFFFu);                    // tile_dim1 (tile_dim2=0)
    g1[5] = (int)stride_elems;                          // tensor_dim0_stride[31:0]
    g1[6] = 0;                                          // stride hi / dim1_stride lo
    g1[7] = 0;
    i32x4 z4 = (i32x4)0;
    i32x8 z8 = (i32x8)0;
    __builtin_amdgcn_tensor_load_to_lds(g0, g1, z4, z4, z8, 0);
}

// Build a 16-element bf16 fragment from two contiguous 8-element (16B) LDS
// chunks.  Chunk base addresses are 16B aligned (rows padded to 16B multiple).
__device__ __forceinline__ bf16x16 load_frag(const __bf16* row, int o0, int o1) {
    bf16x8 lo = *(const __attribute__((address_space(3))) bf16x8*)
                    ((const __attribute__((address_space(3))) __bf16*)row + o0);
    bf16x8 hi = *(const __attribute__((address_space(3))) bf16x8*)
                    ((const __attribute__((address_space(3))) __bf16*)row + o1);
    bf16x16 r;
#pragma unroll
    for (int i = 0; i < 8; ++i) { r[i] = lo[i]; r[i + 8] = hi[i]; }
    return r;
}

__device__ __forceinline__ f32x8 wmma_bf16(bf16x16 a, bf16x16 b, f32x8 c) {
    return __builtin_amdgcn_wmma_f32_16x16x32_bf16(false, a, false, b,
                                                   (short)0, c, false, false);
}

__device__ __forceinline__ float red_max16(float x) {
    x = fmaxf(x, __shfl_xor(x, 1, 16));
    x = fmaxf(x, __shfl_xor(x, 2, 16));
    x = fmaxf(x, __shfl_xor(x, 4, 16));
    x = fmaxf(x, __shfl_xor(x, 8, 16));
    return x;
}
__device__ __forceinline__ float red_sum16(float x) {
    x += __shfl_xor(x, 1, 16);
    x += __shfl_xor(x, 2, 16);
    x += __shfl_xor(x, 4, 16);
    x += __shfl_xor(x, 8, 16);
    return x;
}

// ---------------------------------------------------------------------------
// GEMM + bias:  Y[m][n] = sum_k X[m][k] * W[n][k] + bias[n]
// Block = 256 threads (8 waves), tile 256(M) x 64(N), K-step 64, DOUBLE-
// BUFFERED: tile it+1 is staged (TDM issue / manual converts) while the
// WMMAs consume tile it.  One barrier per K-step.
// IN_BF16:  X is bf16 row-major -> staged by the TDM (no VALU, no VMEM).
// OUT_BF16: Y written as bf16 (feeds attention / next GEMM).
// ---------------------------------------------------------------------------
#define GT_M 256
#define GT_N 64
#define GT_K 64

template <bool IN_BF16, bool OUT_BF16>
__global__ __launch_bounds__(256)
void mha_gemm_bias(const void* __restrict__ Xv, const float* __restrict__ W,
                   const float* __restrict__ bias, void* __restrict__ Yv,
                   int M, int N, int K)
{
    __shared__ __align__(16) __bf16 Xs[2][GT_M][GT_K + 8];   // 2 x 36 KB
    __shared__ __align__(16) __bf16 Ws[2][GT_N][GT_K + 8];   // 2 x  9 KB

    const int tid  = threadIdx.x;
    const int wave = tid >> 5;
    const int lane = tid & 31;
    const int half = lane >> 4;
    const int ln   = lane & 15;

    const int mbase = blockIdx.y * GT_M;
    const int nbase = blockIdx.x * GT_N;
    const int K4  = K >> 2;
    const int NIT = K / GT_K;

    auto stage = [&](int it, int buf) {
        const int k0 = it * GT_K;
        if constexpr (IN_BF16) {
            if (wave == 0)
                tdm_load_2d_bf16(lds_off(&Xs[buf][0][0]),
                                 (const __bf16*)Xv + (size_t)mbase * K + k0,
                                 GT_K, GT_M, (unsigned)K);
        } else {
            const f32x4* X4 = (const f32x4*)((const float*)Xv + (size_t)mbase * K + k0);
#pragma unroll
            for (int i = 0; i < 16; ++i) {
                int idx = tid + i * 256;          // 0..4095
                int r = idx >> 4, c = idx & 15;
                f32x4 xv = X4[(size_t)r * K4 + c];
                *(bf16x4*)&Xs[buf][r][c * 4] = cvt4(xv);
            }
        }
        const f32x4* W4 = (const f32x4*)(W + (size_t)nbase * K + k0);
#pragma unroll
        for (int i = 0; i < 4; ++i) {
            int idx = tid + i * 256;              // 0..1023
            int r = idx >> 4, c = idx & 15;
            f32x4 wv = W4[(size_t)r * K4 + c];
            *(bf16x4*)&Ws[buf][r][c * 4] = cvt4(wv);
        }
    };

    f32x8 acc[2][4];
#pragma unroll
    for (int g = 0; g < 2; ++g)
#pragma unroll
        for (int t = 0; t < 4; ++t) acc[g][t] = (f32x8)(0.0f);

    stage(0, 0);
    for (int it = 0; it < NIT; ++it) {
        const int buf = it & 1;
        if constexpr (IN_BF16) {
            if (wave == 0) __builtin_amdgcn_s_wait_tensorcnt(0);
        }
        __syncthreads();                 // publish tile `it`
        if (it + 1 < NIT) stage(it + 1, buf ^ 1);   // overlap with compute

#pragma unroll
        for (int kk = 0; kk < GT_K; kk += 32) {
            bf16x16 af0 = load_frag(&Xs[buf][wave * 32 + ln][kk],      half * 8, 16 + half * 8);
            bf16x16 af1 = load_frag(&Xs[buf][wave * 32 + 16 + ln][kk], half * 8, 16 + half * 8);
#pragma unroll
            for (int t = 0; t < 4; ++t) {
                bf16x16 bf = load_frag(&Ws[buf][t * 16 + ln][kk], half * 8, 16 + half * 8);
                acc[0][t] = wmma_bf16(af0, bf, acc[0][t]);
                acc[1][t] = wmma_bf16(af1, bf, acc[1][t]);
            }
        }
    }

    // epilogue: C layout -> row = r + 8*half, col = ln (per 16-wide tile)
#pragma unroll
    for (int g = 0; g < 2; ++g)
#pragma unroll
    for (int t = 0; t < 4; ++t) {
        int n = nbase + t * 16 + ln;
        float bv = bias[n];
#pragma unroll
        for (int r = 0; r < 8; ++r) {
            int m = mbase + wave * 32 + g * 16 + r + 8 * half;
            float y = acc[g][t][r] + bv;
            if constexpr (OUT_BF16)
                ((__bf16*)Yv)[(size_t)m * N + n] = f2bf(y);
            else
                ((float*)Yv)[(size_t)m * N + n] = y;
        }
    }
}

// ---------------------------------------------------------------------------
// Flash attention:  per block = (batch*head, 64-query tile), 128 threads.
// K tiles DMA'd by the TDM, V tiles manually transposed into LDS; both
// double-buffered so block kb+1 streams in while WMMAs consume block kb.
// ---------------------------------------------------------------------------
__global__ __launch_bounds__(128)
void mha_flash_attn(const __bf16* __restrict__ Qh, const __bf16* __restrict__ Kh,
                    const __bf16* __restrict__ Vh, __bf16* __restrict__ O)
{
    __shared__ __align__(16) __bf16 Qs[64][72];          // [q][d]   144B rows
    __shared__ __align__(16) __bf16 Ks[2][64][72];       // [j][d]
    __shared__ __align__(16) __bf16 Vst[2][64][72];      // [d][j]  (transposed)
    __shared__ __align__(16) __bf16 Ps[4][16][72];       // wave-private P tiles

    const int tid  = threadIdx.x;
    const int wave = tid >> 5;
    const int lane = tid & 31;
    const int half = lane >> 4;
    const int ln   = lane & 15;

    const int bh = blockIdx.y;                 // 0..47
    const int b  = bh / NUM_HEADS;
    const int h  = bh % NUM_HEADS;
    const int qt = blockIdx.x;                 // 0..31 (64-row query tiles)

    const size_t headoff = (size_t)b * SEQ * D_MODEL + (size_t)h * D_K;
    const __bf16* Qb = Qh + headoff;
    const __bf16* Kb = Kh + headoff;
    const __bf16* Vb = Vh + headoff;
    const int NB = SEQ / 64;

    auto stageKV = [&](int kb, int buf) {
        if (wave == 0)
            tdm_load_2d_bf16(lds_off(&Ks[buf][0][0]),
                             Kb + (size_t)(kb * 64) * D_MODEL,
                             64, 64, D_MODEL);
        // V tile: bf16 b128 loads + LDS transpose scatter (4 chunks / thread)
#pragma unroll
        for (int i = 0; i < 4; ++i) {
            int idx = tid + i * 128;              // 0..511
            int r = idx >> 3, c = idx & 7;        // row j, 8-elem chunk
            bf16x8 vv = *(const bf16x8*)(Vb + (size_t)(kb * 64 + r) * D_MODEL + c * 8);
#pragma unroll
            for (int e = 0; e < 8; ++e)
                Vst[buf][c * 8 + e][r] = vv[e];
        }
    };

    // prologue: Q tile + first K tile DMAs in flight together
    if (wave == 0)
        tdm_load_2d_bf16(lds_off(&Qs[0][0]),
                         Qb + (size_t)(qt * 64) * D_MODEL,
                         64, 64, D_MODEL);
    stageKV(0, 0);

    float m_run[8], l_run[8];
#pragma unroll
    for (int r = 0; r < 8; ++r) { m_run[r] = -INFINITY; l_run[r] = 0.0f; }
    f32x8 o_acc[4];
#pragma unroll
    for (int t = 0; t < 4; ++t) o_acc[t] = (f32x8)(0.0f);

    // softmax in base-2 domain: p = exp2(s*scale*log2e - m2)
    const float SCL = 0.125f * 1.44269504088896340736f;   // 1/sqrt(64) * log2(e)

    for (int kb = 0; kb < NB; ++kb) {
        const int buf = kb & 1;
        if (wave == 0) __builtin_amdgcn_s_wait_tensorcnt(0);
        __syncthreads();                       // publish K/V (and Q on kb==0)
        if (kb + 1 < NB) stageKV(kb + 1, buf ^ 1);   // overlap with compute

        // ---- scores S = Q(16x64) @ K^T(64x64): 4 tiles x 2 WMMA ----
        f32x8 s[4];
#pragma unroll
        for (int t = 0; t < 4; ++t) s[t] = (f32x8)(0.0f);

        bf16x16 qf0 = load_frag(&Qs[wave * 16 + ln][0],  half * 8, 16 + half * 8);
        bf16x16 qf1 = load_frag(&Qs[wave * 16 + ln][32], half * 8, 16 + half * 8);
#pragma unroll
        for (int t = 0; t < 4; ++t) {
            bf16x16 kf0 = load_frag(&Ks[buf][t * 16 + ln][0],  half * 8, 16 + half * 8);
            bf16x16 kf1 = load_frag(&Ks[buf][t * 16 + ln][32], half * 8, 16 + half * 8);
            s[t] = wmma_bf16(qf0, kf0, s[t]);
            s[t] = wmma_bf16(qf1, kf1, s[t]);
        }

        // ---- online softmax over the 16x64 score strip ----
#pragma unroll
        for (int t = 0; t < 4; ++t) s[t] = s[t] * SCL;
#pragma unroll
        for (int r = 0; r < 8; ++r) {
            float loc = fmaxf(fmaxf(s[0][r], s[1][r]), fmaxf(s[2][r], s[3][r]));
            loc = red_max16(loc);
            float mnew  = fmaxf(m_run[r], loc);
            float alpha = exp2f(m_run[r] - mnew);      // first iter: exp2(-inf)=0
            float lsum = 0.0f;
#pragma unroll
            for (int t = 0; t < 4; ++t) {
                float p = exp2f(s[t][r] - mnew);
                s[t][r] = p;
                lsum += p;
            }
            lsum = red_sum16(lsum);
            l_run[r] = l_run[r] * alpha + lsum;
            m_run[r] = mnew;
#pragma unroll
            for (int t = 0; t < 4; ++t) o_acc[t][r] *= alpha;
        }

        // ---- P -> wave-private LDS (C layout -> [m][j]) ----
#pragma unroll
        for (int t = 0; t < 4; ++t)
#pragma unroll
            for (int r = 0; r < 8; ++r)
                Ps[wave][r + 8 * half][t * 16 + ln] = f2bf(s[t][r]);
        // same-wave DS ops are in-order: no barrier needed before re-reading Ps

        // ---- O += P(16x64) @ V(64x64): 4 d-tiles x 2 WMMA ----
        bf16x16 pf0 = load_frag(&Ps[wave][ln][0],  half * 8, 16 + half * 8);
        bf16x16 pf1 = load_frag(&Ps[wave][ln][32], half * 8, 16 + half * 8);
#pragma unroll
        for (int dt = 0; dt < 4; ++dt) {
            bf16x16 vf0 = load_frag(&Vst[buf][dt * 16 + ln][0],  half * 8, 16 + half * 8);
            bf16x16 vf1 = load_frag(&Vst[buf][dt * 16 + ln][32], half * 8, 16 + half * 8);
            o_acc[dt] = wmma_bf16(pf0, vf0, o_acc[dt]);
            o_acc[dt] = wmma_bf16(pf1, vf1, o_acc[dt]);
        }
    }

    // ---- normalize rows and store bf16 [B*S, D_MODEL] ----
#pragma unroll
    for (int r = 0; r < 8; ++r) {
        float inv = 1.0f / l_run[r];
        int m = qt * 64 + wave * 16 + r + 8 * half;
        size_t rowbase = (size_t)(b * SEQ + m) * D_MODEL + (size_t)h * D_K;
#pragma unroll
        for (int t = 0; t < 4; ++t)
            O[rowbase + t * 16 + ln] = f2bf(o_acc[t][r] * inv);
    }
}

// ---------------------------------------------------------------------------
// Launch: 3 projection GEMMs (bf16 out) -> flash attention (bf16 in/out)
//         -> output GEMM (TDM bf16 in, f32 out).
// Workspace: 4 * 8192*768 bf16 = 50 MB.
// ---------------------------------------------------------------------------
extern "C" void kernel_launch(void* const* d_in, const int* in_sizes, int n_in,
                              void* d_out, int out_size, void* d_ws, size_t ws_size,
                              hipStream_t stream) {
    const float* q   = (const float*)d_in[0];
    const float* k   = (const float*)d_in[1];
    const float* v   = (const float*)d_in[2];
    const float* w_q = (const float*)d_in[3];
    const float* b_q = (const float*)d_in[4];
    const float* w_k = (const float*)d_in[5];
    const float* b_k = (const float*)d_in[6];
    const float* w_v = (const float*)d_in[7];
    const float* b_v = (const float*)d_in[8];
    const float* w_o = (const float*)d_in[9];
    const float* b_o = (const float*)d_in[10];

    __bf16* wsb = (__bf16*)d_ws;
    const size_t per = (size_t)MROWS * D_MODEL;
    __bf16* Qh = wsb;
    __bf16* Kh = wsb + per;
    __bf16* Vh = wsb + 2 * per;
    __bf16* AO = wsb + 3 * per;

    dim3 ggrid(D_MODEL / GT_N, MROWS / GT_M);   // (12, 32)
    mha_gemm_bias<false, true><<<ggrid, 256, 0, stream>>>(q, w_q, b_q, Qh,
                                                          MROWS, D_MODEL, D_MODEL);
    mha_gemm_bias<false, true><<<ggrid, 256, 0, stream>>>(k, w_k, b_k, Kh,
                                                          MROWS, D_MODEL, D_MODEL);
    mha_gemm_bias<false, true><<<ggrid, 256, 0, stream>>>(v, w_v, b_v, Vh,
                                                          MROWS, D_MODEL, D_MODEL);

    dim3 agrid(SEQ / 64, BATCH * NUM_HEADS);    // (32, 48)
    mha_flash_attn<<<agrid, 128, 0, stream>>>(Qh, Kh, Vh, AO);

    mha_gemm_bias<true, false><<<ggrid, 256, 0, stream>>>(AO, w_o, b_o, d_out,
                                                          MROWS, D_MODEL, D_MODEL);
}
